// SimpleCNNWithSABlock_37512244363926
// MI455X (gfx1250) — matrix operand order repaired and checked by reading, not compiled, not run on hardware
//
#include <hip/hip_runtime.h>

typedef __attribute__((ext_vector_type(16))) _Float16 v16h;
typedef __attribute__((ext_vector_type(8)))  float    v8f;

static constexpr int BATCH = 2;
static constexpr int C_IN  = 3;
static constexpr int IMG   = 128;
static constexpr int HID   = 64;
static constexpr int HEADS = 4;
static constexpr int DIMH  = 16;
static constexpr int PP    = 64;        // pooled size
static constexpr int SS    = PP * PP;   // 4096 tokens
static constexpr float ATTN_SCALE = 0.25f;   // 16^-0.5
static constexpr float BN_EPS = 1e-5f;

// ---------------- conv3x3 + bias, accumulate per-channel sum/sumsq ----------------
__global__ void conv_stats_kernel(const float* __restrict__ x, const float* __restrict__ w,
                                  const float* __restrict__ bias, float* __restrict__ y,
                                  float* __restrict__ chsum, float* __restrict__ chsumsq) {
    int b = blockIdx.x / (HID * IMG);
    int o = (blockIdx.x / IMG) % HID;
    int h = blockIdx.x % IMG;
    int wx = threadIdx.x;                         // 0..127 = output column
    float acc = bias[o];
    for (int ci = 0; ci < C_IN; ++ci) {
        const float* xp = x + ((b * C_IN + ci) * IMG) * IMG;
        const float* wp = w + ((o * C_IN + ci) * 3) * 3;
        #pragma unroll
        for (int kh = 0; kh < 3; ++kh) {
            int ih = h - 1 + kh;
            if (ih < 0 || ih >= IMG) continue;
            #pragma unroll
            for (int kw = 0; kw < 3; ++kw) {
                int iw = wx - 1 + kw;
                if (iw < 0 || iw >= IMG) continue;
                acc += xp[ih * IMG + iw] * wp[kh * 3 + kw];
            }
        }
    }
    y[((b * HID + o) * IMG + h) * IMG + wx] = acc;
    __shared__ float s1[128], s2[128];
    s1[wx] = acc; s2[wx] = acc * acc;
    __syncthreads();
    for (int st = 64; st > 0; st >>= 1) {
        if (wx < st) { s1[wx] += s1[wx + st]; s2[wx] += s2[wx + st]; }
        __syncthreads();
    }
    if (wx == 0) { atomicAdd(&chsum[o], s1[0]); atomicAdd(&chsumsq[o], s2[0]); }
}

__global__ void bn_finalize_kernel(const float* __restrict__ chsum, const float* __restrict__ chsumsq,
                                   const float* __restrict__ gamma, const float* __restrict__ beta,
                                   float* __restrict__ scale, float* __restrict__ shift) {
    int c = threadIdx.x;
    float n  = (float)(BATCH * IMG * IMG);
    float mu = chsum[c] / n;
    float var = chsumsq[c] / n - mu * mu;
    float r  = rsqrtf(var + BN_EPS);
    float sc = gamma[c] * r;
    scale[c] = sc;
    shift[c] = beta[c] - mu * sc;
}

// ---------------- BN + ReLU + 2x2 maxpool -> tokens (B,S,64) f16 ----------------
__global__ void bn_relu_pool_kernel(const float* __restrict__ y, const float* __restrict__ scale,
                                    const float* __restrict__ shift, _Float16* __restrict__ t) {
    int gid = blockIdx.x * blockDim.x + threadIdx.x;        // ((b*64+c)*64+ph)*64+pw
    if (gid >= BATCH * HID * PP * PP) return;
    int pw = gid & (PP - 1);
    int ph = (gid >> 6) & (PP - 1);
    int c  = (gid >> 12) & (HID - 1);
    int b  = gid >> 18;
    const float* yp = y + ((b * HID + c) * IMG + 2 * ph) * IMG + 2 * pw;
    float sc = scale[c], sh = shift[c];
    float v0 = fmaxf(yp[0]       * sc + sh, 0.f);
    float v1 = fmaxf(yp[1]       * sc + sh, 0.f);
    float v2 = fmaxf(yp[IMG]     * sc + sh, 0.f);
    float v3 = fmaxf(yp[IMG + 1] * sc + sh, 0.f);
    float m = fmaxf(fmaxf(v0, v1), fmaxf(v2, v3));
    int s = ph * PP + pw;
    t[(b * SS + s) * HID + c] = (_Float16)m;
}

__global__ void f32_to_f16_kernel(const float* __restrict__ in, _Float16* __restrict__ out, int n) {
    int i = blockIdx.x * blockDim.x + threadIdx.x;
    if (i < n) out[i] = (_Float16)in[i];
}

// ---------------- QKV GEMM via WMMA: (B*S,64) x (64,192) ----------------
// Q,K stored (b,h,S,16) row-major f16; V stored transposed (b,h,16,S) f16.
// `which` (q/k/v) and `head` are wave-uniform: derived from nbase only.
__global__ __launch_bounds__(128) void qkv_gemm_kernel(
        const _Float16* __restrict__ t, const _Float16* __restrict__ wq,
        const float* __restrict__ qkv_b,
        _Float16* __restrict__ Qb, _Float16* __restrict__ Kb, _Float16* __restrict__ Vt) {
    int wave = (blockIdx.x * blockDim.x + threadIdx.x) >> 5;
    int lane = threadIdx.x & 31;
    int mt = wave % 512;            // 512 M-tiles of 16 rows
    int nt = wave / 512;            // 12 N-tiles of 16 cols
    if (nt >= 12) return;
    int mbase = mt * 16, nbase = nt * 16;
    int lm = lane & 15;
    bool hi = lane >= 16;

    v8f acc = {};
    #pragma unroll
    for (int ks = 0; ks < 2; ++ks) {
        union { v16h v; uint4 q[2]; unsigned u[8]; } A, B;
        int row   = mbase + lm;
        int cbase = ks * 32 + (hi ? 8 : 0);
        A.q[0] = *(const uint4*)(t + row * HID + cbase);        // K=cbase..+7
        A.q[1] = *(const uint4*)(t + row * HID + cbase + 16);   // K=cbase+16..+23
        int n  = nbase + lm;
        int wb = ks * 32 + (hi ? 8 : 0);
        B.q[0] = *(const uint4*)(wq + n * HID + wb);
        B.q[1] = *(const uint4*)(wq + n * HID + wb + 16);
        acc = __builtin_amdgcn_wmma_f32_16x16x32_f16(false, A.v, false, B.v,
                                                     (short)0, acc, false, false);
    }
    float bn = qkv_b[nbase + lm];
    int which = nbase >> 6;              // wave-uniform: 0=q 1=k 2=v
    int head  = (nbase >> 4) & 3;        // wave-uniform
    _Float16* __restrict__ dstQK = (which == 0) ? Qb : Kb;
    #pragma unroll
    for (int i = 0; i < 8; ++i) {
        int m = mbase + i + (hi ? 8 : 0);
        int b = m >> 12, s = m & (SS - 1);
        _Float16 val = (_Float16)(acc[i] + bn);
        int bh = b * HEADS + head;
        if (which < 2) dstQK[(bh * SS + s) * DIMH + lm] = val;     // coalesced rows
        else           Vt[(bh * DIMH + lm) * SS + s]    = val;     // transposed store
    }
}

// ---------------- decomposed rel-pos tables: relh/relw (b,h,S,64) ----------------
__global__ void relpos_kernel(const _Float16* __restrict__ Qb, const float* __restrict__ rph,
                              const float* __restrict__ rpw, float* __restrict__ relh,
                              float* __restrict__ relw) {
    int gid = blockIdx.x * blockDim.x + threadIdx.x;     // (bh*S + s)*64 + k
    if (gid >= BATCH * HEADS * SS * PP) return;
    int k  = gid & 63;
    int s  = (gid >> 6) & (SS - 1);
    int bh = gid >> 18;
    int qh = s >> 6, qw = s & 63;
    const _Float16* qp = Qb + (bh * SS + s) * DIMH;
    const float* rh = rph + (qh - k + 63) * DIMH;
    const float* rw = rpw + (qw - k + 63) * DIMH;
    float ah = 0.f, aw = 0.f;
    #pragma unroll
    for (int c = 0; c < DIMH; ++c) {
        float qv = (float)qp[c];
        ah += qv * rh[c];
        aw += qv * rw[c];
    }
    relh[gid] = ah;
    relw[gid] = aw;
}

// ---------------- flash attention with decomposed rel-pos bias ----------------
// One wave per (b, h, 16-query tile). Online softmax; P@V via WMMA accumulate.
// 1/sqrt(d) is pre-folded into the Q A-operand registers (rel tables use raw Q).
// Output: column sums (over all queries) accumulated into featsum (B,64).
__global__ __launch_bounds__(128) void attn_kernel(
        const _Float16* __restrict__ Qb, const _Float16* __restrict__ Kb,
        const _Float16* __restrict__ Vt, const float* __restrict__ relh,
        const float* __restrict__ relw, float* __restrict__ featsum) {
    __shared__ float    lrel[4][2][16][64];   // per-wave 16x64 relh/relw tiles
    __shared__ _Float16 lp[4][16][32];        // per-wave P staging (16 q x 32 keys)

    int wv   = threadIdx.x >> 5;
    int lane = threadIdx.x & 31;
    int tile = blockIdx.x * 4 + wv;           // 0..2047
    int qt   = tile & 255;
    int bh   = tile >> 8;
    int b    = bh >> 2, head = bh & 3;
    int qbase = qt * 16;
    int lm   = lane & 15;
    bool hi  = lane >= 16;
    int mof  = hi ? 8 : 0;

    // stage rel-pos bias tiles to LDS (reused 128x)
    for (int idx = lane; idx < 16 * 64; idx += 32) {
        int r = idx >> 6, c = idx & 63;
        lrel[wv][0][r][c] = relh[(bh * SS + qbase + r) * 64 + c];
        lrel[wv][1][r][c] = relw[(bh * SS + qbase + r) * 64 + c];
    }

    // Q as A-operand (16x32, K padded: dims 0..15 real, 16..31 zero), scale folded in
    union { v16h v; uint4 q[2]; unsigned u[8]; _Float16 h[16]; } Aq;
    Aq.q[0] = *(const uint4*)(Qb + (bh * SS + qbase + lm) * DIMH + (hi ? 8 : 0));
    Aq.u[4] = Aq.u[5] = Aq.u[6] = Aq.u[7] = 0u;
    #pragma unroll
    for (int j = 0; j < 8; ++j) Aq.h[j] = Aq.h[j] * (_Float16)ATTN_SCALE;

    v8f o = {}, l = {}, mrun;
    #pragma unroll
    for (int i = 0; i < 8; ++i) mrun[i] = -__builtin_inff();

    for (int kb = 0; kb < SS; kb += 32) {
        // K tiles as B-operands (dims on contraction axis, padded upper half)
        union { v16h v; uint4 q[2]; } Bk0, Bk1, Bv;
        if (!hi) {
            const _Float16* kp0 = Kb + (bh * SS + kb + lm) * DIMH;
            const _Float16* kp1 = Kb + (bh * SS + kb + 16 + lm) * DIMH;
            Bk0.q[0] = *(const uint4*)kp0; Bk0.q[1] = make_uint4(0,0,0,0);
            Bk1.q[0] = *(const uint4*)kp1; Bk1.q[1] = make_uint4(0,0,0,0);
            if (kb + 32 < SS)
                __builtin_prefetch(Kb + (bh * SS + kb + 32 + lm) * DIMH, 0, 1);
        } else {
            Bk0.q[0] = *(const uint4*)(Kb + (bh * SS + kb + lm) * DIMH + 8); Bk0.q[1] = make_uint4(0,0,0,0);
            Bk1.q[0] = *(const uint4*)(Kb + (bh * SS + kb + 16 + lm) * DIMH + 8); Bk1.q[1] = make_uint4(0,0,0,0);
        }
        // V tile as B-operand: rows = 32 keys, cols = 16 dims (Vt is (bh,dim,S))
        const _Float16* vp = Vt + (bh * DIMH + lm) * SS + kb + (hi ? 8 : 0);
        Bv.q[0] = *(const uint4*)vp; Bv.q[1] = *(const uint4*)(vp + 16);
        if (kb + 32 < SS)
            __builtin_prefetch(vp + 32, 0, 1);

        v8f z = {};
        v8f s0 = __builtin_amdgcn_wmma_f32_16x16x32_f16(false, Aq.v, false, Bk0.v,
                                                        (short)0, z, false, false);
        v8f s1 = __builtin_amdgcn_wmma_f32_16x16x32_f16(false, Aq.v, false, Bk1.v,
                                                        (short)0, z, false, false);

        int key0 = kb + lm, key1 = kb + 16 + lm;
        int kh0 = key0 >> 6, kw0 = key0 & 63, kh1 = key1 >> 6, kw1 = key1 & 63;
        #pragma unroll
        for (int i = 0; i < 8; ++i) {
            int m = i + mof;
            s0[i] = s0[i] + lrel[wv][0][m][kh0] + lrel[wv][1][m][kw0];
            s1[i] = s1[i] + lrel[wv][0][m][kh1] + lrel[wv][1][m][kw1];
        }
        // online softmax: per-row (16-lane butterfly) max & sum
        #pragma unroll
        for (int i = 0; i < 8; ++i) {
            float t = fmaxf(s0[i], s1[i]);
            t = fmaxf(t, __shfl_xor(t, 1, 32));
            t = fmaxf(t, __shfl_xor(t, 2, 32));
            t = fmaxf(t, __shfl_xor(t, 4, 32));
            t = fmaxf(t, __shfl_xor(t, 8, 32));
            float mn = fmaxf(mrun[i], t);
            float alpha = __expf(mrun[i] - mn);
            float e0 = __expf(s0[i] - mn);
            float e1 = __expf(s1[i] - mn);
            float r = e0 + e1;
            r += __shfl_xor(r, 1, 32);
            r += __shfl_xor(r, 2, 32);
            r += __shfl_xor(r, 4, 32);
            r += __shfl_xor(r, 8, 32);
            l[i] = l[i] * alpha + r;
            o[i] = o[i] * alpha;
            mrun[i] = mn;
            lp[wv][i + mof][lm]      = (_Float16)e0;   // C-layout -> LDS
            lp[wv][i + mof][16 + lm] = (_Float16)e1;
        }
        asm volatile("s_wait_dscnt 0" ::: "memory");   // P store -> P reload (same wave)
        // reload P in A-operand layout (16 queries x 32 keys, full K)
        union { v16h v; unsigned u[8]; } Ap;
        {
            const unsigned* base = (const unsigned*)&lp[wv][lm][0];
            int lo2 = hi ? 4 : 0;                       // half-offset/2
            #pragma unroll
            for (int j = 0; j < 4; ++j) {
                Ap.u[j]     = base[lo2 + j];            // keys (0|8)+2j
                Ap.u[4 + j] = base[8 + lo2 + j];        // keys 16+(0|8)+2j
            }
        }
        o = __builtin_amdgcn_wmma_f32_16x16x32_f16(false, Ap.v, false, Bv.v,
                                                   (short)0, o, false, false);
    }
    // normalize rows, accumulate column sums (mean commutes with out-proj)
    float partial = 0.f;
    #pragma unroll
    for (int i = 0; i < 8; ++i) { partial += o[i] * __frcp_rn(l[i]); }
    atomicAdd(&featsum[b * HID + head * DIMH + lm], partial);
}

// ---------------- tail: mean -> out-proj -> fc ----------------
__global__ void head_kernel(const float* __restrict__ featsum, const float* __restrict__ out_w,
                            const float* __restrict__ out_b, const float* __restrict__ fc_w,
                            const float* __restrict__ fc_b, float* __restrict__ out) {
    int tid = threadIdx.x;
    if (tid >= BATCH * 2) return;
    int b = tid >> 1, n = tid & 1;
    float invS = 1.f / (float)SS;
    float acc = fc_b[n];
    for (int j = 0; j < HID; ++j) {
        float po = out_b[j];
        for (int c = 0; c < HID; ++c)
            po += featsum[b * HID + c] * invS * out_w[j * HID + c];
        acc += fc_w[n * HID + j] * po;
    }
    out[b * 2 + n] = acc;
}

extern "C" void kernel_launch(void* const* d_in, const int* in_sizes, int n_in,
                              void* d_out, int out_size, void* d_ws, size_t ws_size,
                              hipStream_t stream) {
    (void)in_sizes; (void)n_in; (void)out_size; (void)ws_size;
    const float* x        = (const float*)d_in[0];
    const float* conv_w   = (const float*)d_in[1];
    const float* conv_b   = (const float*)d_in[2];
    const float* bn_gamma = (const float*)d_in[3];
    const float* bn_beta  = (const float*)d_in[4];
    const float* qkv_w    = (const float*)d_in[5];
    const float* qkv_b    = (const float*)d_in[6];
    const float* rph      = (const float*)d_in[7];
    const float* rpw      = (const float*)d_in[8];
    const float* out_w    = (const float*)d_in[9];
    const float* out_b    = (const float*)d_in[10];
    const float* fc_w     = (const float*)d_in[11];
    const float* fc_b     = (const float*)d_in[12];
    float* out = (float*)d_out;

    char* ws = (char*)d_ws;
    size_t off = 0;
    auto alloc = [&](size_t bytes) { size_t r = off; off += (bytes + 255) & ~(size_t)255; return r; };

    float*    y       = (float*)   (ws + alloc(sizeof(float) * BATCH * HID * IMG * IMG));
    float*    stats   = (float*)   (ws + alloc(sizeof(float) * 256)); // chsum64|chsumsq64|featsum128
    float*    chsum   = stats;
    float*    chsumsq = stats + 64;
    float*    featsum = stats + 128;
    float*    bnscale = (float*)   (ws + alloc(sizeof(float) * HID));
    float*    bnshift = (float*)   (ws + alloc(sizeof(float) * HID));
    _Float16* t16     = (_Float16*)(ws + alloc(sizeof(_Float16) * BATCH * SS * HID));
    _Float16* w16     = (_Float16*)(ws + alloc(sizeof(_Float16) * 3 * HID * HID));
    _Float16* Qb      = (_Float16*)(ws + alloc(sizeof(_Float16) * BATCH * HEADS * SS * DIMH));
    _Float16* Kb      = (_Float16*)(ws + alloc(sizeof(_Float16) * BATCH * HEADS * SS * DIMH));
    _Float16* Vt      = (_Float16*)(ws + alloc(sizeof(_Float16) * BATCH * HEADS * SS * DIMH));
    float*    relh    = (float*)   (ws + alloc(sizeof(float) * BATCH * HEADS * SS * PP));
    float*    relw    = (float*)   (ws + alloc(sizeof(float) * BATCH * HEADS * SS * PP));

    hipMemsetAsync(stats, 0, 256 * sizeof(float), stream);

    conv_stats_kernel<<<BATCH * HID * IMG, 128, 0, stream>>>(x, conv_w, conv_b, y, chsum, chsumsq);
    bn_finalize_kernel<<<1, 64, 0, stream>>>(chsum, chsumsq, bn_gamma, bn_beta, bnscale, bnshift);
    bn_relu_pool_kernel<<<(BATCH * HID * PP * PP) / 256, 256, 0, stream>>>(y, bnscale, bnshift, t16);
    f32_to_f16_kernel<<<(3 * HID * HID + 255) / 256, 256, 0, stream>>>(qkv_w, w16, 3 * HID * HID);
    qkv_gemm_kernel<<<1536, 128, 0, stream>>>(t16, w16, qkv_b, Qb, Kb, Vt);
    relpos_kernel<<<(BATCH * HEADS * SS * PP) / 256, 256, 0, stream>>>(Qb, rph, rpw, relh, relw);
    attn_kernel<<<512, 128, 0, stream>>>(Qb, Kb, Vt, relh, relw, featsum);
    head_kernel<<<1, 32, 0, stream>>>(featsum, out_w, out_b, fc_w, fc_b, out);
}
